// SeparableConv_36386962932352
// MI455X (gfx1250) — matrix operand order repaired and checked by reading, not compiled, not run on hardware
//
#include <hip/hip_runtime.h>

typedef _Float16 h16;
typedef __attribute__((ext_vector_type(16))) _Float16 v16h;
typedef __attribute__((ext_vector_type(8)))  _Float16 v8h;
typedef __attribute__((ext_vector_type(4)))  _Float16 v4h;
typedef __attribute__((ext_vector_type(2)))  _Float16 v2h;
typedef __attribute__((ext_vector_type(8)))  float    v8f;

#define HH 112
#define WW 112
#define CI 128
#define CO 256
#define LSTR 136            // padded LDS row stride in f16 (272B: 16B aligned, bank-shift 4)
#define NTHREADS 224        // 7 waves -> 7 x 16 = 112 M rows

__global__ __launch_bounds__(NTHREADS)
void sepconv_wmma(const float* __restrict__ x,
                  const float* __restrict__ dwk,
                  const float* __restrict__ dwb,
                  const float* __restrict__ pwk,
                  const float* __restrict__ pwb,
                  float* __restrict__ out)
{
    // Total static LDS: (128 + 112) * 136 * 2 = 65,280 B  (<= 64KB CU-mode limit)
    __shared__ h16 sB[128 * LSTR];   // half of pointwise weights, [co_local][k]
    __shared__ h16 sY[WW  * LSTR];   // depthwise output (f16), [w][k]

    const int t = threadIdx.x;
    const int h = blockIdx.x;        // 0..111
    const int n = blockIdx.y;        // 0..31

    // ---------------- stage 1: depthwise 3x3 (fp32) -> sY (f16) ----------------
    // 112 pixels * 32 channel-groups (float4) = 3584 items / 224 threads = 16 iters
    for (int idx = t; idx < WW * (CI / 4); idx += NTHREADS) {
        const int cg = idx & 31;
        const int w  = idx >> 5;
        const int c  = cg * 4;
        float4 acc = *(const float4*)&dwb[c];
        #pragma unroll
        for (int kh = 0; kh < 3; ++kh) {
            const int hh = h + kh - 1;
            if (hh < 0 || hh >= HH) continue;
            #pragma unroll
            for (int kw = 0; kw < 3; ++kw) {
                const int ww = w + kw - 1;
                if (ww < 0 || ww >= WW) continue;
                const float4 xv = *(const float4*)&x[(((size_t)n * HH + hh) * WW + ww) * CI + c];
                const float4 kv = *(const float4*)&dwk[(kh * 3 + kw) * CI + c];
                acc.x = fmaf(xv.x, kv.x, acc.x);
                acc.y = fmaf(xv.y, kv.y, acc.y);
                acc.z = fmaf(xv.z, kv.z, acc.z);
                acc.w = fmaf(xv.w, kv.w, acc.w);
            }
        }
        v4h y4 = { (h16)acc.x, (h16)acc.y, (h16)acc.z, (h16)acc.w };
        *(v4h*)&sY[w * LSTR + c] = y4;   // 8B-aligned -> ds_store_b64
    }

    const int lane = t & 31;
    const int wid  = t >> 5;          // 0..6
    const int m0   = wid * 16;        // pixel (M) base of this wave's subtile
    const int half = lane >> 4;       // lane half selects K sub-block per ISA layout
    const int l16  = lane & 15;

    // ---------------- stage 2: pointwise GEMM, N in two halves of 128 ----------------
    for (int nh = 0; nh < 2; ++nh) {
        __syncthreads();   // sY ready (nh==0) / previous half's sB reads done (nh==1)

        // stage pw weight half: f32 [k][co] (global) -> f16 [co_local][k] (LDS)
        // idx -> co fastest: coalesced global reads; k pairs -> one 4B LDS store
        for (int idx = t; idx < 128 * (CI / 2); idx += NTHREADS) {
            const int col = idx & 127;        // co_local
            const int kp  = idx >> 7;         // k pair 0..63
            const int k   = kp * 2;
            const int co  = nh * 128 + col;
            v2h p = { (h16)pwk[(size_t)k * CO + co],
                      (h16)pwk[(size_t)(k + 1) * CO + co] };
            *(v2h*)&sB[col * LSTR + k] = p;
        }
        __syncthreads();

        v8f acc[8];
        const v8f vzero = {0.f, 0.f, 0.f, 0.f, 0.f, 0.f, 0.f, 0.f};
        #pragma unroll
        for (int i = 0; i < 8; ++i) acc[i] = vzero;

        #pragma unroll
        for (int kk = 0; kk < 4; ++kk) {
            const int k0 = kk * 32;
            // A fragment (16x32 f16): VGPR0-3 = K[k0+8h .. +8), VGPR4-7 = K[k0+16+8h .. +8)
            const h16* ap = &sY[(m0 + l16) * LSTR + k0 + half * 8];
            const v8h alo = *(const v8h*)(ap);        // ds_load_b128
            const v8h ahi = *(const v8h*)(ap + 16);   // ds_load_b128
            v16h a;
            #pragma unroll
            for (int i = 0; i < 8; ++i) { a[i] = alo[i]; a[i + 8] = ahi[i]; }

            #pragma unroll
            for (int nt = 0; nt < 8; ++nt) {
                // B fragment (32x16 f16): elems 0-7 = K[k0+8h .. +8), elems 8-15 = K[k0+16+8h .. +8)
                const h16* bp = &sB[(nt * 16 + l16) * LSTR + k0 + half * 8];
                const v8h blo = *(const v8h*)(bp);
                const v8h bhi = *(const v8h*)(bp + 16);
                v16h b;
                #pragma unroll
                for (int i = 0; i < 8; ++i) { b[i] = blo[i]; b[i + 8] = bhi[i]; }

                acc[nt] = __builtin_amdgcn_wmma_f32_16x16x32_f16(
                    false, a, false, b, (short)0, acc[nt], false, false);
            }
        }

        // ---------------- epilogue: bias + non-temporal store ----------------
        float* orow = out + ((size_t)n * HH + h) * WW * (size_t)CO;
        #pragma unroll
        for (int nt = 0; nt < 8; ++nt) {
            const int co = nh * 128 + nt * 16 + l16;
            const float bias = pwb[co];
            #pragma unroll
            for (int r = 0; r < 8; ++r) {
                const int w = m0 + half * 8 + r;   // C/D layout: VGPR r -> M = r + 8*half
                __builtin_nontemporal_store(acc[nt][r] + bias,
                                            &orow[(size_t)w * CO + co]);
            }
        }
    }
}

extern "C" void kernel_launch(void* const* d_in, const int* in_sizes, int n_in,
                              void* d_out, int out_size, void* d_ws, size_t ws_size,
                              hipStream_t stream) {
    (void)in_sizes; (void)n_in; (void)out_size; (void)d_ws; (void)ws_size;
    const float* x   = (const float*)d_in[0];
    const float* dwk = (const float*)d_in[1];
    const float* dwb = (const float*)d_in[2];
    const float* pwk = (const float*)d_in[3];
    const float* pwb = (const float*)d_in[4];
    float* out = (float*)d_out;

    dim3 grid(HH, 32);   // one block per (h, n) image row: 3584 blocks
    sepconv_wmma<<<grid, dim3(NTHREADS), 0, stream>>>(x, dwk, dwb, pwk, pwb, out);
}
